// MolecularMPNN_39779987095913
// MI455X (gfx1250) — matrix operand order repaired and checked by reading, not compiled, not run on hardware
//
#include <hip/hip_runtime.h>
#include <hip/hip_bf16.h>
#include <math.h>

// ---------------- problem constants (from reference) ----------------
#define N_NODES   100000
#define N_EDGES   1600000
#define N_GRAPHS  4096
#define F_IN      64
#define HID       128
#define HEADS     4
#define CPH       32
#define NLAYERS   3
#define NTASKS    8
#define NEG_SLOPE 0.2f
#define E_TOT     (N_EDGES + N_NODES)   // 1,700,000

typedef __attribute__((ext_vector_type(16))) __bf16   v16bf;
typedef __attribute__((ext_vector_type(8)))  float    v8f;
typedef __attribute__((ext_vector_type(4)))  unsigned u32x4;

static inline int cdiv_h(long long a, long long b) { return (int)((a + b - 1) / b); }

// fp32 -> bf16 (round-to-nearest-even)
__device__ __forceinline__ __bf16 f2bf(float f) {
    union { float f; unsigned u; } v; v.f = f;
    unsigned r = v.u + 0x7FFFu + ((v.u >> 16) & 1u);
    unsigned short s = (unsigned short)(r >> 16);
    return __builtin_bit_cast(__bf16, s);
}

// two 16B chunks -> one v16bf fragment
struct Frag32 { u32x4 lo, hi; };
__device__ __forceinline__ v16bf load_frag(const __bf16* p0, const __bf16* p1) {
    Frag32 t;
    t.lo = *(const u32x4*)p0;
    t.hi = *(const u32x4*)p1;
    return __builtin_bit_cast(v16bf, t);
}

__device__ __forceinline__ void atomicMaxFloat(float* addr, float val) {
    unsigned* ua = (unsigned*)addr;
    unsigned old = *ua;
    for (;;) {
        float cur = __uint_as_float(old);
        if (cur >= val) break;
        unsigned assumed = old;
        old = atomicCAS(ua, assumed, __float_as_uint(val));
        if (old == assumed) break;
    }
}

// ---------------- fills / converts / packs ----------------
__global__ void fill_kernel(float* __restrict__ p, long long n, float v) {
    long long i = (long long)blockIdx.x * blockDim.x + threadIdx.x;
    if (i < n) p[i] = v;
}

__global__ void f2bf_kernel(const float* __restrict__ in, __bf16* __restrict__ out, long long n) {
    long long i = (long long)blockIdx.x * blockDim.x + threadIdx.x;
    if (i < n) out[i] = f2bf(in[i]);
}

// Pack fp32 B[K,N] (row-major) into WMMA lane order, bf16:
// Bp[ (tk*(N/16)+tn)*512 + lane*16 + j ] = B[ tk*32 + (lane>>4)*8 + (j&7) + (j>>3)*16 ][ tn*16 + (lane&15) ]
__global__ void pack_b_kernel(const float* __restrict__ B0, __bf16* __restrict__ Bp0,
                              int K, int N, long long strideB, long long strideBp)
{
    const int t = blockIdx.z;
    const float* B  = B0  + (long long)t * strideB;
    __bf16*      Bp = Bp0 + (long long)t * strideBp;
    int idx = blockIdx.x * blockDim.x + threadIdx.x;
    if (idx >= K * N) return;
    const int j      = idx & 15;
    const int lane   = (idx >> 4) & 31;
    const int tile   = idx >> 9;
    const int tilesN = N >> 4;
    const int tn = tile % tilesN, tk = tile / tilesN;
    const int kk = tk * 32 + (lane >> 4) * 8 + (j & 7) + (j >> 3) * 16;
    const int nn = tn * 16 + (lane & 15);
    Bp[idx] = f2bf(B[(long long)kk * N + nn]);
}

// ---------------- WMMA bf16 GEMM: C[M,N] = A[M,K] @ B[K,N] (+bias)(+relu) ----------------
// A: bf16 row-major. Bp: bf16 pre-packed (pack_b_kernel). C: fp32.
// One wave32 computes a 32x32 C block (2x2 WMMA tiles). Requires M%32==0, N%32==0, K%32==0.
template <bool HAS_BIAS, bool RELU>
__global__ void wmma_gemm_kernel(const __bf16* __restrict__ A,
                                 const __bf16* __restrict__ Bp0,
                                 const float* __restrict__ bias0,
                                 float* __restrict__ C0,
                                 int M, int K, int N,
                                 long long strideBp, long long strideBias, long long strideC)
{
    const int t = blockIdx.z;
    const __bf16* Bp   = Bp0 + (long long)t * strideBp;
    const float*  bias = HAS_BIAS ? (bias0 + (long long)t * strideBias) : nullptr;
    float*        C    = C0 + (long long)t * strideC;

    const int wave   = threadIdx.x >> 5;
    const int lane   = threadIdx.x & 31;
    const int half   = lane >> 4;
    const int lane16 = lane & 15;

    const int blkN = N >> 5;
    const long long numBlk = (long long)(M >> 5) * blkN;
    const long long blk    = (long long)blockIdx.x * (blockDim.x >> 5) + wave;
    if (blk >= numBlk) return;
    const int m0 = (int)(blk / blkN) * 32;
    const int n0 = (int)(blk % blkN) * 32;
    const int tilesN = N >> 4;

    v8f acc00 = {}, acc01 = {}, acc10 = {}, acc11 = {};
    for (int k0 = 0; k0 < K; k0 += 32) {
        // A fragments: lane holds row m0(+16)+lane16, K = k0+half*8+[0..7] and +16
        const __bf16* ar0 = A + (long long)(m0 + lane16) * K + k0 + half * 8;
        const __bf16* ar1 = ar0 + (long long)16 * K;
        v16bf a0 = load_frag(ar0, ar0 + 16);
        v16bf a1 = load_frag(ar1, ar1 + 16);
        // B fragments: packed; lane's 16 values contiguous
        const __bf16* bp = Bp + ((long long)(k0 >> 5) * tilesN + (n0 >> 4)) * 512 + lane * 16;
        v16bf b0 = load_frag(bp, bp + 8);
        v16bf b1 = load_frag(bp + 512, bp + 520);

        acc00 = __builtin_amdgcn_wmma_f32_16x16x32_bf16(false, a0, false, b0, (short)0, acc00, false, false);
        acc01 = __builtin_amdgcn_wmma_f32_16x16x32_bf16(false, a0, false, b1, (short)0, acc01, false, false);
        acc10 = __builtin_amdgcn_wmma_f32_16x16x32_bf16(false, a1, false, b0, (short)0, acc10, false, false);
        acc11 = __builtin_amdgcn_wmma_f32_16x16x32_bf16(false, a1, false, b1, (short)0, acc11, false, false);
    }

    const int   col0 = n0 + lane16, col1 = col0 + 16;
    const float bv0  = HAS_BIAS ? bias[col0] : 0.0f;
    const float bv1  = HAS_BIAS ? bias[col1] : 0.0f;
#pragma unroll
    for (int r = 0; r < 8; ++r) {
        const int row0 = m0 + half * 8 + r;
        const int row1 = row0 + 16;
        float v00 = acc00[r] + bv0, v01 = acc01[r] + bv1;
        float v10 = acc10[r] + bv0, v11 = acc11[r] + bv1;
        if (RELU) {
            v00 = fmaxf(v00, 0.f); v01 = fmaxf(v01, 0.f);
            v10 = fmaxf(v10, 0.f); v11 = fmaxf(v11, 0.f);
        }
        C[(long long)row0 * N + col0] = v00;
        C[(long long)row0 * N + col1] = v01;
        C[(long long)row1 * N + col0] = v10;
        C[(long long)row1 * N + col1] = v11;
    }
}

// ---------------- per-node attention logits ----------------
__global__ void alpha_kernel(const float* __restrict__ hW,
                             const float* __restrict__ att_s,
                             const float* __restrict__ att_d,
                             float* __restrict__ as_, float* __restrict__ ad_)
{
    int idx = blockIdx.x * blockDim.x + threadIdx.x;
    if (idx >= N_NODES * HEADS) return;
    const int node = idx >> 2, head = idx & 3;
    const float* v = hW + (long long)node * HID + head * CPH;
    const float* s = att_s + head * CPH;
    const float* d = att_d + head * CPH;
    float a = 0.f, b = 0.f;
#pragma unroll
    for (int i = 0; i < CPH; ++i) { a += v[i] * s[i]; b += v[i] * d[i]; }
    as_[idx] = a; ad_[idx] = b;
}

// ---------------- edge pass 1: segment max ----------------
__global__ void edge_max_kernel(const int* __restrict__ ei,
                                const float* __restrict__ as_,
                                const float* __restrict__ ad_,
                                float* __restrict__ m)
{
    long long e = (long long)blockIdx.x * blockDim.x + threadIdx.x;
    if (e >= E_TOT) return;
    int src, dst;
    if (e < N_EDGES) { src = ei[e]; dst = ei[N_EDGES + e]; }
    else             { src = dst = (int)(e - N_EDGES); }
#pragma unroll
    for (int h = 0; h < HEADS; ++h) {
        float v = as_[src * HEADS + h] + ad_[dst * HEADS + h];
        v = v > 0.f ? v : NEG_SLOPE * v;
        atomicMaxFloat(&m[dst * HEADS + h], v);
    }
}

// ---------------- edge pass 2: fused exp + denom + message scatter ----------------
__global__ void edge_accum_kernel(const int* __restrict__ ei,
                                  const float* __restrict__ as_,
                                  const float* __restrict__ ad_,
                                  const float* __restrict__ m,
                                  const float* __restrict__ hW,
                                  float* __restrict__ denom,
                                  float* __restrict__ agg)
{
    const int lane = threadIdx.x & 31;
    long long e = ((long long)blockIdx.x * blockDim.x + threadIdx.x) >> 5;
    if (e >= E_TOT) return;
    int src, dst;
    if (e < N_EDGES) { src = ei[e]; dst = ei[N_EDGES + e]; }
    else             { src = dst = (int)(e - N_EDGES); }
    const float* hrow = hW  + (long long)src * HID;
    float*       arow = agg + (long long)dst * HID;
#pragma unroll
    for (int h = 0; h < HEADS; ++h) {
        float v = as_[src * HEADS + h] + ad_[dst * HEADS + h];
        v = v > 0.f ? v : NEG_SLOPE * v;
        float p = __expf(v - m[dst * HEADS + h]);
        if (lane == h) atomicAdd(&denom[dst * HEADS + h], p);
        atomicAdd(&arow[h * CPH + lane], p * hrow[h * CPH + lane]);
    }
}

// ---------------- per-node finalize: h = relu(agg/denom + bias); writes fp32 + bf16 ----------------
__global__ void node_finalize_kernel(const float* __restrict__ agg,
                                     const float* __restrict__ denom,
                                     const float* __restrict__ bias,
                                     float* __restrict__ h,
                                     __bf16* __restrict__ hbf)
{
    long long idx = (long long)blockIdx.x * blockDim.x + threadIdx.x;
    if (idx >= (long long)N_NODES * HID) return;
    const int node = (int)(idx >> 7);
    const int ch   = (int)(idx & 127);
    const float d  = denom[node * HEADS + (ch >> 5)];
    const float v  = fmaxf(agg[idx] / d + bias[ch], 0.0f);
    h[idx]   = v;
    hbf[idx] = f2bf(v);
}

// ---------------- global mean pool ----------------
__global__ void pool_accum_kernel(const float* __restrict__ h,
                                  const int* __restrict__ batch,
                                  float* __restrict__ sums,
                                  float* __restrict__ cnts)
{
    long long idx = (long long)blockIdx.x * blockDim.x + threadIdx.x;
    if (idx >= (long long)N_NODES * HID) return;
    const int node = (int)(idx >> 7);
    const int ch   = (int)(idx & 127);
    const int g    = batch[node];
    atomicAdd(&sums[(long long)g * HID + ch], h[idx]);
    if (ch == 0) atomicAdd(&cnts[g], 1.0f);
}

__global__ void pool_div_kernel(const float* __restrict__ sums,
                                const float* __restrict__ cnts,
                                __bf16* __restrict__ pooled_bf)
{
    int idx = blockIdx.x * blockDim.x + threadIdx.x;
    if (idx >= N_GRAPHS * HID) return;
    pooled_bf[idx] = f2bf(sums[idx] / fmaxf(cnts[idx >> 7], 1.0f));
}

// ---------------- per-task head output ----------------
__global__ void head_out_kernel(const float* __restrict__ z,     // [T, G, 64]
                                const float* __restrict__ W2,    // [T, 64]
                                const float* __restrict__ b2,    // [T]
                                float* __restrict__ out)         // [G, T]
{
    int idx = blockIdx.x * blockDim.x + threadIdx.x;
    if (idx >= N_GRAPHS * NTASKS) return;
    const int g = idx >> 3, t = idx & 7;
    const float* zp = z + ((long long)t * N_GRAPHS + g) * (HID / 2);
    const float* wp = W2 + t * (HID / 2);
    float acc = b2[t];
#pragma unroll
    for (int k = 0; k < HID / 2; ++k) acc += zp[k] * wp[k];
    out[idx] = acc;
}

// ==================================================================
extern "C" void kernel_launch(void* const* d_in, const int* in_sizes, int n_in,
                              void* d_out, int out_size, void* d_ws, size_t ws_size,
                              hipStream_t stream)
{
    (void)in_sizes; (void)n_in; (void)out_size; (void)ws_size;
    const float* x     = (const float*)d_in[0];
    const int*   ei    = (const int*)  d_in[1];
    const int*   batch = (const int*)  d_in[2];
    const float* enc_W = (const float*)d_in[3];
    const float* enc_b = (const float*)d_in[4];
    const float* lin_W = (const float*)d_in[5];
    const float* att_s = (const float*)d_in[6];
    const float* att_d = (const float*)d_in[7];
    const float* gat_b = (const float*)d_in[8];
    const float* hW1   = (const float*)d_in[9];
    const float* hb1   = (const float*)d_in[10];
    const float* hW2   = (const float*)d_in[11];
    const float* hb2   = (const float*)d_in[12];
    float* out = (float*)d_out;

    // workspace bump allocator (256B aligned)
    char*  base = (char*)d_ws;
    size_t off  = 0;
    auto alloc = [&](size_t bytes) -> void* {
        void* p = base + off;
        off = (off + bytes + 255) & ~(size_t)255;
        return p;
    };
    const long long NE = (long long)N_NODES * HID;
    float*  h        = (float*) alloc(NE * 4);
    float*  hw       = (float*) alloc(NE * 4);
    float*  agg      = (float*) alloc(NE * 4);
    __bf16* h_bf     = (__bf16*)alloc(NE * 2);
    __bf16* x_bf     = (__bf16*)alloc((long long)N_NODES * F_IN * 2);
    float*  as_      = (float*) alloc((long long)N_NODES * HEADS * 4);
    float*  ad_      = (float*) alloc((long long)N_NODES * HEADS * 4);
    float*  mmax     = (float*) alloc((long long)N_NODES * HEADS * 4);
    float*  denom    = (float*) alloc((long long)N_NODES * HEADS * 4);
    float*  sums     = (float*) alloc((long long)N_GRAPHS * HID * 4);
    float*  cnts     = (float*) alloc((long long)N_GRAPHS * 4);
    __bf16* pooledbf = (__bf16*)alloc((long long)N_GRAPHS * HID * 2);
    float*  z        = (float*) alloc((long long)NTASKS * N_GRAPHS * (HID / 2) * 4);
    __bf16* encBp    = (__bf16*)alloc((long long)F_IN * HID * 2);
    __bf16* linBp    = (__bf16*)alloc((long long)HID * HID * 2);
    __bf16* headBp   = (__bf16*)alloc((long long)NTASKS * HID * (HID / 2) * 2);

    const int BLK = 256;   // 8 wave32s

    // ---- one-shot conversions/packs ----
    f2bf_kernel<<<cdiv_h((long long)N_NODES * F_IN, BLK), BLK, 0, stream>>>(
        x, x_bf, (long long)N_NODES * F_IN);
    pack_b_kernel<<<dim3(cdiv_h((long long)F_IN * HID, BLK), 1, 1), BLK, 0, stream>>>(
        enc_W, encBp, F_IN, HID, 0, 0);
    pack_b_kernel<<<dim3(cdiv_h((long long)HID * (HID / 2), BLK), 1, NTASKS), BLK, 0, stream>>>(
        hW1, headBp, HID, HID / 2, (long long)HID * (HID / 2), (long long)HID * (HID / 2));

    // ---- encoder: h = x @ enc_W + enc_b ----
    {
        long long blks = (long long)(N_NODES / 32) * (HID / 32);
        wmma_gemm_kernel<true, false><<<dim3(cdiv_h(blks, 8), 1, 1), BLK, 0, stream>>>(
            x_bf, encBp, enc_b, h, N_NODES, F_IN, HID, 0, 0, 0);
        f2bf_kernel<<<cdiv_h(NE, BLK), BLK, 0, stream>>>(h, h_bf, NE);
    }

    // ---- 3 GAT layers ----
    for (int l = 0; l < NLAYERS; ++l) {
        fill_kernel<<<cdiv_h((long long)N_NODES * HEADS, BLK), BLK, 0, stream>>>(
            mmax, (long long)N_NODES * HEADS, -INFINITY);
        fill_kernel<<<cdiv_h((long long)N_NODES * HEADS, BLK), BLK, 0, stream>>>(
            denom, (long long)N_NODES * HEADS, 0.0f);
        fill_kernel<<<cdiv_h(NE, BLK), BLK, 0, stream>>>(agg, NE, 0.0f);

        pack_b_kernel<<<dim3(cdiv_h((long long)HID * HID, BLK), 1, 1), BLK, 0, stream>>>(
            lin_W + (long long)l * HID * HID, linBp, HID, HID, 0, 0);

        long long blks = (long long)(N_NODES / 32) * (HID / 32);
        wmma_gemm_kernel<false, false><<<dim3(cdiv_h(blks, 8), 1, 1), BLK, 0, stream>>>(
            h_bf, linBp, nullptr, hw, N_NODES, HID, HID, 0, 0, 0);

        alpha_kernel<<<cdiv_h((long long)N_NODES * HEADS, BLK), BLK, 0, stream>>>(
            hw, att_s + l * HEADS * CPH, att_d + l * HEADS * CPH, as_, ad_);

        edge_max_kernel<<<cdiv_h((long long)E_TOT, BLK), BLK, 0, stream>>>(ei, as_, ad_, mmax);

        edge_accum_kernel<<<cdiv_h((long long)E_TOT * 32, BLK), BLK, 0, stream>>>(
            ei, as_, ad_, mmax, hw, denom, agg);

        node_finalize_kernel<<<cdiv_h(NE, BLK), BLK, 0, stream>>>(
            agg, denom, gat_b + l * HID, h, h_bf);
    }

    // ---- global mean pool ----
    fill_kernel<<<cdiv_h((long long)N_GRAPHS * HID, BLK), BLK, 0, stream>>>(
        sums, (long long)N_GRAPHS * HID, 0.0f);
    fill_kernel<<<cdiv_h((long long)N_GRAPHS, BLK), BLK, 0, stream>>>(
        cnts, (long long)N_GRAPHS, 0.0f);
    pool_accum_kernel<<<cdiv_h(NE, BLK), BLK, 0, stream>>>(h, batch, sums, cnts);
    pool_div_kernel<<<cdiv_h((long long)N_GRAPHS * HID, BLK), BLK, 0, stream>>>(
        sums, cnts, pooledbf);

    // ---- task heads layer 1: z[t] = relu(pooled @ W1[t] + b1[t]) ----
    {
        long long blks = (long long)(N_GRAPHS / 32) * ((HID / 2) / 32);
        wmma_gemm_kernel<true, true><<<dim3(cdiv_h(blks, 8), 1, NTASKS), BLK, 0, stream>>>(
            pooledbf, headBp, hb1, z, N_GRAPHS, HID, HID / 2,
            (long long)HID * (HID / 2), (long long)(HID / 2),
            (long long)N_GRAPHS * (HID / 2));
    }

    // ---- task heads layer 2 + transpose to [G,T] ----
    head_out_kernel<<<cdiv_h((long long)N_GRAPHS * NTASKS, BLK), BLK, 0, stream>>>(
        z, hW2, hb2, out);
}
